// ArcFace_49821620634120
// MI455X (gfx1250) — compile-verified, matrix operand order checked by reading, not run on hardware
//
#include <hip/hip_runtime.h>

#define Nn 1024
#define Dd 512
#define Cc 100000
#define BM 128
#define BN 128
#define BK 32
#define BKP 40                          // padded row stride (bf16): 80B = 20 banks -> conflict-free
#define Sc 30.0f
#define COSM 0.87758256189037271612f    // cos(0.5)
#define SINM 0.47942553860420300027f    // sin(0.5)

typedef __attribute__((ext_vector_type(16))) __bf16 v16bf;
typedef __attribute__((ext_vector_type(8)))  float  v8f;

union Frag { v16bf v; uint4 q[2]; };

static __device__ __forceinline__ unsigned short f2bf(float f) {
  unsigned int u = __float_as_uint(f);
  u = (u + 0x7FFFu + ((u >> 16) & 1u)) >> 16;   // round-to-nearest-even
  return (unsigned short)u;
}

// One block per row: L2-normalize a 512-float row, emit packed bf16.
__global__ void __launch_bounds__(256)
normalize_rows(const float* __restrict__ in, unsigned short* __restrict__ out) {
  const int row = blockIdx.x;
  const int t = threadIdx.x;
  const float2 v = ((const float2*)(in + (size_t)row * Dd))[t];
  float ss = v.x * v.x + v.y * v.y;
  #pragma unroll
  for (int off = 16; off; off >>= 1) ss += __shfl_xor(ss, off, 32);
  __shared__ float red[8];
  if ((t & 31) == 0) red[t >> 5] = ss;
  __syncthreads();
  float tot = 0.0f;
  #pragma unroll
  for (int i = 0; i < 8; ++i) tot += red[i];
  const float inv = 1.0f / fmaxf(sqrtf(tot), 1e-12f);
  const unsigned int p = (unsigned int)f2bf(v.x * inv) |
                         ((unsigned int)f2bf(v.y * inv) << 16);
  ((unsigned int*)(out + (size_t)row * Dd))[t] = p;
}

// 128x128 tile GEMM (bf16 WMMA, f32 acc), double-buffered LDS, margin epilogue.
__global__ void __launch_bounds__(256)
arcface_gemm(const unsigned short* __restrict__ A,   // e_bf16 [N,D]
             const unsigned short* __restrict__ B,   // w_bf16 [C,D]
             const int* __restrict__ labels,
             float* __restrict__ out) {
  __shared__ unsigned short As[2][BM * BKP];
  __shared__ unsigned short Bs[2][BN * BKP];

  // M-tiles on the FAST grid dimension: the 8 blocks sharing a weight tile
  // are dispatched adjacently -> weight tile is served from L2, not HBM.
  const int mBase = blockIdx.x * BM;
  const int nBase = blockIdx.y * BN;
  const int t    = threadIdx.x;
  const int lane = t & 31;
  const int wid  = t >> 5;
  const int wm   = wid >> 1;               // 0..3 -> 32 rows each
  const int wn   = wid & 1;                // 0..1 -> 64 cols each
  const int lrow  = lane & 15;
  const int halfk = (lane < 16) ? 0 : 8;   // K-chunk select per ISA layout

  const v8f vzero = {};
  v8f acc[2][4];
  #pragma unroll
  for (int i = 0; i < 2; ++i)
    #pragma unroll
    for (int j = 0; j < 4; ++j) acc[i][j] = vzero;

  // Stage one 128x32 bf16 tile of A and B into LDS buffer `buf`.
  // Out-of-range weight rows are CLAMPED (not branched): the duplicated row
  // is computed but never stored (epilogue guards gn < Cc). This keeps the
  // staging branch-free so loads batch under counted loadcnt waits.
  auto load_tile = [&](int kt, int buf) {
    #pragma unroll
    for (int i = 0; i < 2; ++i) {
      const int idx = t + i * 256;          // 0..511
      const int row = idx >> 2;
      const int q   = idx & 3;
      const unsigned short* ap = A + (size_t)(mBase + row) * Dd + kt + q * 8;
      const int brow = min(nBase + row, Cc - 1);
      const unsigned short* bp = B + (size_t)brow * Dd + kt + q * 8;
      const uint4 av = *(const uint4*)ap;
      const uint4 bv = *(const uint4*)bp;
      *(uint4*)(&As[buf][row * BKP + q * 8]) = av;
      *(uint4*)(&Bs[buf][row * BKP + q * 8]) = bv;
      if (kt + BK < Dd) {                   // prefetch the tile after this one
        __builtin_prefetch(ap + BK, 0, 1);
        __builtin_prefetch(bp + BK, 0, 1);
      }
    }
  };

  load_tile(0, 0);
  __syncthreads();

  int buf = 0;
  for (int kt = 0; kt < Dd; kt += BK, buf ^= 1) {
    if (kt + BK < Dd) load_tile(kt + BK, buf ^ 1);  // overlaps with WMMAs below

    // Fragments per ISA 16-bit layouts:
    // lanes 0-15: K 0-7 & 16-23; lanes 16-31: K 8-15 & 24-31; M/N = lane%16.
    Frag af[2], bfr[4];
    #pragma unroll
    for (int fm = 0; fm < 2; ++fm) {
      const unsigned short* p = &As[buf][(wm * 32 + fm * 16 + lrow) * BKP + halfk];
      af[fm].q[0] = *(const uint4*)(p);
      af[fm].q[1] = *(const uint4*)(p + 16);
    }
    #pragma unroll
    for (int fn = 0; fn < 4; ++fn) {
      const unsigned short* p = &Bs[buf][(wn * 64 + fn * 16 + lrow) * BKP + halfk];
      bfr[fn].q[0] = *(const uint4*)(p);
      bfr[fn].q[1] = *(const uint4*)(p + 16);
    }

    #pragma unroll
    for (int fm = 0; fm < 2; ++fm)
      #pragma unroll
      for (int fn = 0; fn < 4; ++fn)
        acc[fm][fn] = __builtin_amdgcn_wmma_f32_16x16x32_bf16(
            false, af[fm].v, false, bfr[fn].v, (short)0, acc[fm][fn],
            false, false);

    __syncthreads();   // next-buffer stores done AND this buffer fully read
  }

  // Epilogue: clip, angular margin at label column, scale by S.
  #pragma unroll
  for (int fm = 0; fm < 2; ++fm) {
    #pragma unroll
    for (int r = 0; r < 8; ++r) {
      const int gm  = mBase + wm * 32 + fm * 16 + r + ((lane < 16) ? 0 : 8);
      const int lbl = labels[gm];
      #pragma unroll
      for (int fn = 0; fn < 4; ++fn) {
        const int gn = nBase + wn * 64 + fn * 16 + lrow;
        if (gn < Cc) {
          float v = acc[fm][fn][r];
          v = fminf(fmaxf(v, -1.0f), 1.0f);
          if (gn == lbl) {
            const float s = sqrtf(fmaxf(1.0f - v * v, 0.0f));
            v = v * COSM - s * SINM;
          }
          out[(size_t)gm * Cc + gn] = Sc * v;
        }
      }
    }
  }
}

extern "C" void kernel_launch(void* const* d_in, const int* in_sizes, int n_in,
                              void* d_out, int out_size, void* d_ws, size_t ws_size,
                              hipStream_t stream) {
  const float* emb    = (const float*)d_in[0];
  const float* wt     = (const float*)d_in[1];
  const int*   labels = (const int*)d_in[2];

  unsigned short* e_bf = (unsigned short*)d_ws;                 // 1 MB
  unsigned short* w_bf = e_bf + (size_t)Nn * Dd;                // +102.4 MB

  normalize_rows<<<Nn, 256, 0, stream>>>(emb, e_bf);
  normalize_rows<<<Cc, 256, 0, stream>>>(wt, w_bf);

  dim3 grid(Nn / BM, (Cc + BN - 1) / BN);                       // 8 x 782, M fastest
  arcface_gemm<<<grid, 256, 0, stream>>>(e_bf, w_bf, labels, (float*)d_out);
}
